// MyMlpLayer_88923002896558
// MI455X (gfx1250) — matrix operand order repaired and checked by reading, not compiled, not run on hardware
//
#include <hip/hip_runtime.h>
#include <hip/hip_bf16.h>
#include <math.h>
#include <stdint.h>

// ---------------------------------------------------------------------------
// MI455X (gfx1250, wave32).
//   GRU scan: 512 sequential steps; per step dual GEMM [128,512]x[512,1536]
//     on bf16 WMMA. Both A-tiles (h, x_t fp32) and the bf16 weight panels
//     stream through LDS via global_load_async_to_lds_b128 (ASYNCcnt,
//     double-buffered slabs) while the matrix pipe consumes the previous
//     slab. r/z gates accumulate x- and h-projections into shared WMMA
//     accumulators (C-chaining) to cut VGPR pressure (no scratch spills).
//   FFN: 3x [32768,512,512] GEMMs, 16x64 strip per wave (4 accumulators),
//     same async-LDS streaming for A and the weight panel.
//   log-softmax + half projection + tiny head.
// ---------------------------------------------------------------------------

typedef __attribute__((ext_vector_type(16))) __bf16 v16bf;
typedef __attribute__((ext_vector_type(8)))  __bf16 v8bf;
typedef __attribute__((ext_vector_type(8)))  float  v8f;

#define WMMA_BF16(a, b, c) \
  __builtin_amdgcn_wmma_f32_16x16x32_bf16(false, (a), false, (b), (short)0, (c), false, false)

// Async 16B global -> LDS (GLOBAL_LOAD_ASYNC_TO_LDS_B128, ASYNCcnt-tracked).
__device__ __forceinline__ void async_g2l_b128(uint32_t ldsoff, const void* gptr) {
  asm volatile("global_load_async_to_lds_b128 %0, %1, off"
               :: "v"(ldsoff), "v"(gptr) : "memory");
}
__device__ __forceinline__ void wait_asynccnt0() {
  asm volatile("s_wait_asynccnt 0x0" ::: "memory");
}
__device__ __forceinline__ uint32_t lds_off(const void* p) {
  return (uint32_t)(uintptr_t)p;   // LDS aperture keeps offset in addr[31:0]
}

// A fragment (16x32 bf16) from LDS-staged fp32 tile [16][32], converted
// in-register. A layout: lane l<16 -> row l, K {0..15};
// lanes 16..31 -> row l-16, K {16..31} (matches B's contiguous K order).
__device__ __forceinline__ v16bf load_a_lds(const float* s, int lane) {
  int row = lane & 15;
  int kb  = (lane >> 4) << 4;
  const float* p = s + row * 32 + kb;
  v16bf a;
#pragma unroll
  for (int i = 0; i < 8; ++i) { a[i] = (__bf16)p[i]; a[i + 8] = (__bf16)p[i + 8]; }
  return a;
}

// B fragment (32x16 bf16) from an LDS-staged weight panel slice laid out as
// [row][32] bf16 (row = output column within the panel). Lane holds column
// n = lane%16 -> panel row tile_base + n, 16 contiguous K values
// (+16 for upper lanes): two ds_load_b128 per lane.
__device__ __forceinline__ v16bf load_b_lds(const __bf16* s, int tile_base, int lane) {
  int row = tile_base + (lane & 15);
  int kb  = (lane >> 4) << 4;
  const v8bf* p = (const v8bf*)(s + row * 32 + kb);
  v8bf lo = p[0], hi = p[1];
  v16bf b;
#pragma unroll
  for (int i = 0; i < 8; ++i) { b[i] = lo[i]; b[i + 8] = hi[i]; }
  return b;
}

__device__ __forceinline__ float sigmoidf_(float x) {
  return 1.0f / (1.0f + __expf(-x));
}

// ---------------------------------------------------------------------------
__global__ void cvt_bf16_kernel(const float* __restrict__ src,
                                __bf16* __restrict__ dst, int n) {
  int i = blockIdx.x * blockDim.x + threadIdx.x;
  if (i < n) dst[i] = (__bf16)src[i];
}

__global__ void zero_kernel(float* __restrict__ p, int n) {
  int i = blockIdx.x * blockDim.x + threadIdx.x;
  if (i < n) p[i] = 0.0f;
}

// ---------------------------------------------------------------------------
// One GRU step. grid = (8, 8), block = 128 (4 waves). Block covers batch
// tile m0=bx*16 and feature range d0blk=by*64; wave w owns d0 = d0blk+w*16.
// Accumulators: acc_r, acc_z (x+h merged via C-chaining), acc_xn, acc_hn.
// ---------------------------------------------------------------------------
__global__ void gru_step_kernel(const float* __restrict__ x0,      // [B,T,D]
                                const float* __restrict__ hprev,   // [B,D]
                                const __bf16* __restrict__ Wih,    // [3D,D]
                                const __bf16* __restrict__ Whh,    // [3D,D]
                                const float* __restrict__ bih,
                                const float* __restrict__ bhh,
                                float* __restrict__ hnext,         // [B,D]
                                float* __restrict__ ebuf,          // [B,T/2,D]
                                int t) {
  const int D = 512, T = 512;
  __shared__ float  sH[2][16 * 32];          // h A-tile, fp32
  __shared__ float  sX[2][16 * 32];          // x_t A-tile, fp32
  __shared__ __bf16 sB[2][6][64 * 32];       // weight panel: slice = mat*3+gate

  int tid  = threadIdx.x;
  int lane = tid & 31;
  int wave = tid >> 5;
  int m0    = blockIdx.x * 16;
  int d0blk = blockIdx.y * 64;
  int dt    = wave * 16;                     // tile base within panel

  // A staging: 128 threads x 16B = one 16x32 fp32 tile
  int srow = tid >> 3, sseg = (tid & 7) << 2;
  const float* hrowp = hprev + (size_t)(m0 + srow) * D + sseg;
  const float* xrowp = x0 + ((size_t)(m0 + srow) * T + t) * D + sseg;

  // B staging: 6 slices x 64 rows x 32 bf16 = 24KB/slab, 12 x 16B per thread.
  // task j: slice s=j>>1 (compile-time), row = (tid>>2) + (j&1)*32, seg=tid&3
  int brow0 = tid >> 2, bseg = (tid & 3) << 3;   // seg offset in bf16 elems

  auto stage_slab = [&](int buf, int k0) {
#pragma unroll
    for (int j = 0; j < 12; ++j) {
      const int s = j >> 1;                      // 0..5 (compile-time)
      const __bf16* base = (s < 3) ? Whh : Wih;
      const int g = (s < 3) ? s : s - 3;
      int row = brow0 + ((j & 1) << 5);
      const __bf16* src = base + (size_t)(g * D + d0blk + row) * D + k0 + bseg;
      async_g2l_b128(lds_off(&sB[buf][s][row * 32 + bseg]), src);
    }
    async_g2l_b128(lds_off(&sH[buf][srow * 32 + sseg]), hrowp + k0);
    async_g2l_b128(lds_off(&sX[buf][srow * 32 + sseg]), xrowp + k0);
  };

  stage_slab(0, 0);
  wait_asynccnt0();
  __syncthreads();

  v8f acc_r = {}, acc_z = {}, acc_xn = {}, acc_hn = {};

  int buf = 0;
#pragma unroll 2
  for (int k0 = 0; k0 < D; k0 += 32, buf ^= 1) {
    if (k0 + 32 < D) stage_slab(buf ^ 1, k0 + 32);   // stream next slab

    v16bf a_h = load_a_lds(&sH[buf][0], lane);
    v16bf a_x = load_a_lds(&sX[buf][0], lane);
    v16bf bh_r = load_b_lds(&sB[buf][0][0], dt, lane);
    v16bf bh_z = load_b_lds(&sB[buf][1][0], dt, lane);
    v16bf bh_n = load_b_lds(&sB[buf][2][0], dt, lane);
    v16bf bx_r = load_b_lds(&sB[buf][3][0], dt, lane);
    v16bf bx_z = load_b_lds(&sB[buf][4][0], dt, lane);
    v16bf bx_n = load_b_lds(&sB[buf][5][0], dt, lane);

    acc_hn = WMMA_BF16(a_h, bh_n, acc_hn);
    acc_xn = WMMA_BF16(a_x, bx_n, acc_xn);
    acc_r  = WMMA_BF16(a_h, bh_r, acc_r);
    acc_z  = WMMA_BF16(a_h, bh_z, acc_z);
    acc_r  = WMMA_BF16(a_x, bx_r, acc_r);    // C-chained: xr+hr in one acc
    acc_z  = WMMA_BF16(a_x, bx_z, acc_z);

    wait_asynccnt0();
    __syncthreads();
  }

  // C/D layout: VGPR i -> row m0 + i + (lane>=16 ? 8 : 0), col d0 + lane%16
  int d = d0blk + dt + (lane & 15);
  int rbase = m0 + ((lane >> 4) << 3);
  float br = bih[d] + bhh[d];
  float bz = bih[D + d] + bhh[D + d];
  float bn_x = bih[2 * D + d], bn_h = bhh[2 * D + d];
#pragma unroll
  for (int i = 0; i < 8; ++i) {
    int b = rbase + i;
    float r = sigmoidf_(acc_r[i] + br);
    float z = sigmoidf_(acc_z[i] + bz);
    float n = tanhf((acc_xn[i] + bn_x) + r * (acc_hn[i] + bn_h));
    float hp = hprev[(size_t)b * D + d];
    float hv = (1.0f - z) * n + z * hp;
    hnext[(size_t)b * D + d] = hv;
    if (t < T / 2) ebuf[((size_t)b * (T / 2) + t) * D + d] = hv;
  }
}

// ---------------------------------------------------------------------------
// GEMM: out[M,N] = act( A[M,K](f32) @ W[N,K](bf16)^T + bias ).
// grid = (M/16, N/256), block = 128 (4 waves), wave -> 16x64 strip
// (4 accumulators). A-tile and 256x32 weight panel stream via async LDS.
// ---------------------------------------------------------------------------
__global__ void gemm_bf16w_kernel(const float* __restrict__ A,
                                  const __bf16* __restrict__ W,
                                  const float* __restrict__ bias,
                                  float* __restrict__ out,
                                  int M, int N, int K, int act) {
  __shared__ float  sA[2][16 * 32];
  __shared__ __bf16 sB[2][256 * 32];

  int tid  = threadIdx.x;
  int lane = tid & 31;
  int wave = tid >> 5;
  int m0    = blockIdx.x * 16;
  int n0blk = blockIdx.y * 256;
  int nt    = wave * 64;                       // wave's strip base in panel

  int srow = tid >> 3, sseg = (tid & 7) << 2;
  const float* arowp = A + (size_t)(m0 + srow) * K + sseg;

  // B staging: 256 rows x 32 bf16 = 16KB/slab, 8 x 16B per thread.
  int brow0 = tid >> 2, bseg = (tid & 3) << 3;

  auto stage_slab = [&](int buf, int k0) {
#pragma unroll
    for (int j = 0; j < 8; ++j) {
      int row = brow0 + (j << 5);
      const __bf16* src = W + (size_t)(n0blk + row) * K + k0 + bseg;
      async_g2l_b128(lds_off(&sB[buf][row * 32 + bseg]), src);
    }
    async_g2l_b128(lds_off(&sA[buf][srow * 32 + sseg]), arowp + k0);
  };

  stage_slab(0, 0);
  wait_asynccnt0();
  __syncthreads();

  v8f acc0 = {}, acc1 = {}, acc2 = {}, acc3 = {};
  int buf = 0;
#pragma unroll 2
  for (int k0 = 0; k0 < K; k0 += 32, buf ^= 1) {
    if (k0 + 32 < K) stage_slab(buf ^ 1, k0 + 32);

    v16bf a  = load_a_lds(&sA[buf][0], lane);
    v16bf b0 = load_b_lds(&sB[buf][0], nt +  0, lane);
    v16bf b1 = load_b_lds(&sB[buf][0], nt + 16, lane);
    v16bf b2 = load_b_lds(&sB[buf][0], nt + 32, lane);
    v16bf b3 = load_b_lds(&sB[buf][0], nt + 48, lane);

    acc0 = WMMA_BF16(a, b0, acc0);
    acc1 = WMMA_BF16(a, b1, acc1);
    acc2 = WMMA_BF16(a, b2, acc2);
    acc3 = WMMA_BF16(a, b3, acc3);

    wait_asynccnt0();
    __syncthreads();
  }

  int colb = lane & 15;
  int rbase = m0 + ((lane >> 4) << 3);
  v8f* accs[4] = {&acc0, &acc1, &acc2, &acc3};
#pragma unroll
  for (int s = 0; s < 4; ++s) {
    int col = n0blk + nt + s * 16 + colb;
    float bv = bias[col];
    v8f& a = *accs[s];
#pragma unroll
    for (int i = 0; i < 8; ++i) {
      float v = a[i] + bv;
      if (act == 1) v = tanhf(v);
      out[(size_t)(rbase + i) * N + col] = v;
    }
  }
}

// ---------------------------------------------------------------------------
__global__ void log_softmax512_kernel(const float* __restrict__ in,
                                      float* __restrict__ out) {
  __shared__ float red[256];
  size_t row = blockIdx.x;
  const float* p = in + row * 512;
  int tid = threadIdx.x;
  float a = p[tid], b = p[tid + 256];

  red[tid] = fmaxf(a, b);
  __syncthreads();
  for (int s = 128; s > 0; s >>= 1) {
    if (tid < s) red[tid] = fmaxf(red[tid], red[tid + s]);
    __syncthreads();
  }
  float mx = red[0];
  __syncthreads();

  red[tid] = __expf(a - mx) + __expf(b - mx);
  __syncthreads();
  for (int s = 128; s > 0; s >>= 1) {
    if (tid < s) red[tid] += red[tid + s];
    __syncthreads();
  }
  float lg = __logf(red[0]);

  float* q = out + row * 512;
  q[tid]       = a - mx - lg;
  q[tid + 256] = b - mx - lg;
}

// ---------------------------------------------------------------------------
__global__ void head7_kernel(const float* __restrict__ H,
                             const float* __restrict__ Wout,
                             const float* __restrict__ bout,
                             float* __restrict__ out, int M) {
  int idx = blockIdx.x * blockDim.x + threadIdx.x;
  int r = idx >> 3, c = idx & 7;
  if (r >= M || c >= 7) return;
  const float* hrow = H + (size_t)r * 256;
  const float* wrow = Wout + c * 256;
  float s = bout[c];
#pragma unroll 4
  for (int j = 0; j < 256; ++j) s += hrow[j] * wrow[j];
  out[(size_t)r * 7 + c] = s;
}

// ---------------------------------------------------------------------------
extern "C" void kernel_launch(void* const* d_in, const int* in_sizes, int n_in,
                              void* d_out, int out_size, void* d_ws, size_t ws_size,
                              hipStream_t stream) {
  const int B = 128, T = 512, D = 512;
  const int M2 = B * (T / 2);

  const float* x0     = (const float*)d_in[0];
  const float* W_ih   = (const float*)d_in[3];
  const float* W_hh   = (const float*)d_in[4];
  const float* b_ih   = (const float*)d_in[5];
  const float* b_hh   = (const float*)d_in[6];
  const float* ffn_W1 = (const float*)d_in[7];
  const float* ffn_b1 = (const float*)d_in[8];
  const float* ffn_W2 = (const float*)d_in[9];
  const float* ffn_b2 = (const float*)d_in[10];
  const float* ffn_W3 = (const float*)d_in[11];
  const float* ffn_b3 = (const float*)d_in[12];
  const float* W_half = (const float*)d_in[13];
  const float* b_half = (const float*)d_in[14];
  const float* W_out  = (const float*)d_in[15];
  const float* b_out  = (const float*)d_in[16];
  float* out = (float*)d_out;

  float* ws      = (float*)d_ws;
  float* hbuf0   = ws;
  float* hbuf1   = hbuf0 + (size_t)B * D;
  float* ebuf    = hbuf1 + (size_t)B * D;
  float* bufA    = ebuf + (size_t)M2 * D;
  float* bufB    = bufA + (size_t)M2 * D;
  __bf16* wih_bf = (__bf16*)(bufB + (size_t)M2 * D);
  __bf16* whh_bf = wih_bf + (size_t)3 * D * D;
  __bf16* w1_bf  = whh_bf + (size_t)3 * D * D;
  __bf16* w2_bf  = w1_bf + (size_t)D * D;
  __bf16* w3_bf  = w2_bf + (size_t)D * D;
  __bf16* wh_bf  = w3_bf + (size_t)D * D;

  {
    int n3 = 3 * D * D, n1 = D * D, nh = (D / 2) * D;
    cvt_bf16_kernel<<<(n3 + 255) / 256, 256, 0, stream>>>(W_ih,   wih_bf, n3);
    cvt_bf16_kernel<<<(n3 + 255) / 256, 256, 0, stream>>>(W_hh,   whh_bf, n3);
    cvt_bf16_kernel<<<(n1 + 255) / 256, 256, 0, stream>>>(ffn_W1, w1_bf,  n1);
    cvt_bf16_kernel<<<(n1 + 255) / 256, 256, 0, stream>>>(ffn_W2, w2_bf,  n1);
    cvt_bf16_kernel<<<(n1 + 255) / 256, 256, 0, stream>>>(ffn_W3, w3_bf,  n1);
    cvt_bf16_kernel<<<(nh + 255) / 256, 256, 0, stream>>>(W_half, wh_bf,  nh);
  }

  zero_kernel<<<(B * D + 255) / 256, 256, 0, stream>>>(hbuf0, B * D);
  {
    float* hp = hbuf0;
    float* hn = hbuf1;
    dim3 grid(B / 16, D / 64);          // (8, 8)
    for (int t = 0; t < T; ++t) {
      gru_step_kernel<<<grid, 128, 0, stream>>>(x0, hp, wih_bf, whh_bf,
                                                b_ih, b_hh, hn, ebuf, t);
      float* tmp = hp; hp = hn; hn = tmp;
    }
  }

  {
    dim3 g512(M2 / 16, D / 256);            // (2048, 2)
    dim3 g256(M2 / 16, (D / 2) / 256);      // (2048, 1)
    gemm_bf16w_kernel<<<g512, 128, 0, stream>>>(ebuf, w1_bf, ffn_b1, bufA, M2, D, D, 1);
    gemm_bf16w_kernel<<<g512, 128, 0, stream>>>(bufA, w2_bf, ffn_b2, bufB, M2, D, D, 1);
    gemm_bf16w_kernel<<<g512, 128, 0, stream>>>(bufB, w3_bf, ffn_b3, bufA, M2, D, D, 0);
    log_softmax512_kernel<<<M2, 256, 0, stream>>>(bufA, bufB);
    gemm_bf16w_kernel<<<g256, 128, 0, stream>>>(bufB, wh_bf, b_half, bufA, M2, D / 2, D, 0);
    head7_kernel<<<(M2 * 8 + 255) / 256, 256, 0, stream>>>(bufA, W_out, b_out, out, M2);
  }
}
